// LSTM_42941083026043
// MI455X (gfx1250) — matrix operand order repaired and checked by reading, not compiled, not run on hardware
//
#include <hip/hip_runtime.h>
#include <hip/hip_bf16.h>

// ---------------------------------------------------------------------------
// Bidirectional ragged-length LSTM for MI455X (gfx1250, wave32, WMMA).
//
// Phase 0: convert X and W to bf16 in workspace.
// Phase 1: Z[dir,t,b,0:4H] = X[b,t,:] @ W_x^T + bias   (parallel bf16 WMMA GEMM)
// Phase 2: per direction, ONE 32-wave workgroup runs the 1024-step recurrence:
//          acc = h @ W_h^T via v_wmma_f32_16x16x32_bf16, gates in f32,
//          c/h state in registers, h exchanged via double-buffered LDS,
//          one barrier per timestep, next-step Z prefetched via global_prefetch.
// ---------------------------------------------------------------------------

typedef __attribute__((ext_vector_type(16))) __bf16 v16bf;
typedef __attribute__((ext_vector_type(8)))  __bf16 v8bf;
typedef __attribute__((ext_vector_type(8)))  float  v8f;

#define Bsz  32
#define Tsz  1024
#define Dsz  256
#define Hsz  256
#define G4H  1024              // 4*H
#define KW   512               // D + H (W row length)

// workspace layout (bytes)
#define WBF_ELEMS (2*G4H*KW)                 // 1,048,576 bf16 (both dirs)
#define XBF_ELEMS (Bsz*Tsz*Dsz)              // 8,388,608 bf16
#define WBF_OFF   0
#define XBF_OFF   ((size_t)WBF_ELEMS*2)      // 2 MiB
#define Z_OFF     (XBF_OFF + (size_t)XBF_ELEMS*2) // + 16 MiB
#define Z_PER_DIR ((size_t)Tsz*Bsz*G4H)      // 33,554,432 floats per dir

// A-fragment (16-bit A 16x32): lane holds 16 contiguous K {c..c+15},
// c = 16*(lane>=16)  ->  two 16-byte chunks at {ko, ko+8} elements.
__device__ __forceinline__ v16bf ldA16(const __bf16* p) {
  union { v16bf v; v8bf h[2]; } u;
  u.h[0] = *(const v8bf*)(p);
  u.h[1] = *(const v8bf*)(p + 8);
  return u.v;
}

// B-fragment (16-bit B 32x16, same K order as A): lane holds column N = lane%16 with
// 16 CONTIGUOUS K values, K base = kbase + 16*(lane>=16) -> one 32-byte load.
__device__ __forceinline__ v16bf ldB16(const __bf16* p) {
  return *(const v16bf*)p;
}

__device__ __forceinline__ v8f wmma_bf16(v16bf a, v16bf b, v8f c) {
  // (neg_a, A, neg_b, B, c_mod, C, reuse_a, reuse_b)
  return __builtin_amdgcn_wmma_f32_16x16x32_bf16(false, a, false, b,
                                                 (short)0, c, false, false);
}

__device__ __forceinline__ float fsigmoid(float x) {
  return 1.0f / (1.0f + __expf(-x));
}
__device__ __forceinline__ float ftanh_(float x) {
  return 2.0f / (1.0f + __expf(-2.0f * x)) - 1.0f;
}

// ---------------------------------------------------------------- Phase 0 ---
__global__ void __launch_bounds__(256) lstm_prep_kernel(
    const float* __restrict__ X, const float* __restrict__ Wf,
    const float* __restrict__ Wb,
    __bf16* __restrict__ xbf, __bf16* __restrict__ wbf)
{
  const int idx    = blockIdx.x * 256 + threadIdx.x;
  const int stride = gridDim.x * 256;
  for (int i = idx; i < XBF_ELEMS; i += stride) xbf[i] = (__bf16)X[i];
  for (int i = idx; i < G4H * KW; i += stride) {
    wbf[i]            = (__bf16)Wf[i];
    wbf[G4H * KW + i] = (__bf16)Wb[i];
  }
}

// ---------------------------------------------------------------- Phase 1 ---
// One wave per 16x16 output tile. M index m = t*32 + b so Z is [T,B,4H].
__global__ void __launch_bounds__(256) lstm_zgemm_kernel(
    const __bf16* __restrict__ xbf, const __bf16* __restrict__ wbf,
    const float* __restrict__ bias_f, const float* __restrict__ bias_b,
    float* __restrict__ Z)
{
  const int lane = threadIdx.x & 31;
  const int job  = blockIdx.x * 8 + (threadIdx.x >> 5); // 262144 tile jobs
  const int nt   = job & 63;
  const int mt   = (job >> 6) & 2047;
  const int dir  = job >> 17;
  const int l16  = lane & 15, hi = lane >> 4;

  const int m  = mt * 16 + l16;              // row = t*32 + b
  const int tt = m >> 5, bb = m & 31;
  const __bf16* arow = xbf + ((size_t)bb * Tsz + tt) * Dsz;
  const int n = nt * 16 + l16;
  const __bf16* brow = wbf + (size_t)dir * (G4H * KW) + (size_t)n * KW; // x-cols

  v8f acc = {};
#pragma unroll
  for (int kc = 0; kc < 8; ++kc) {
    const int koA = kc * 32 + hi * 16;       // A: contiguous 16 K values
    const int koB = kc * 32 + hi * 16;       // B: contiguous 16 K values
    acc = wmma_bf16(ldA16(arow + koA), ldB16(brow + koB), acc);
  }
  const float bn = (dir ? bias_b : bias_f)[n];
  float* zout = Z + (size_t)dir * Z_PER_DIR;
#pragma unroll
  for (int r = 0; r < 8; ++r) {
    const int mr = mt * 16 + hi * 8 + r;     // C layout: M = r + 8*hi
    zout[(size_t)mr * G4H + nt * 16 + l16] = acc[r] + bn;
  }
}

// ---------------------------------------------------------------- Phase 2 ---
// blockIdx.x = direction. 32 waves: wave = (wg<<1)|mt ; wave owns the 16x16
// h-tile (rows mt*16..+15, cols wg*16..+15) and its 4 gate accumulators.
__global__ void __launch_bounds__(1024, 1) lstm_seq_kernel(
    const __bf16* __restrict__ wbf, const float* __restrict__ Z,
    const int* __restrict__ lengths, float* __restrict__ out)
{
  __shared__ alignas(16) __bf16 hbuf[2][32][264]; // +8 bf16 pad: stride 528 B

  const int dir  = blockIdx.x;
  const int tid  = threadIdx.x;
  const int wave = tid >> 5, lane = tid & 31;
  const int mt   = wave & 1, wg = wave >> 1;     // wg in [0,16)
  const int l16  = lane & 15, hi = lane >> 4;

  for (int i = tid; i < 2 * 32 * 264; i += 1024)
    ((__bf16*)hbuf)[i] = (__bf16)0.0f;

  int lenr[8];
#pragma unroll
  for (int r = 0; r < 8; ++r) lenr[r] = lengths[mt * 16 + hi * 8 + r];

  const __bf16* WdR = wbf + (size_t)dir * (G4H * KW) + Dsz; // recurrent cols
  const float*  Zd  = Z + (size_t)dir * Z_PER_DIR;
  const int hcol = wg * 16 + l16;

  int ncol[4];
  const __bf16* bptr[4];
#pragma unroll
  for (int g = 0; g < 4; ++g) {
    ncol[g] = g * Hsz + wg * 16;
    bptr[g] = WdR + (size_t)(ncol[g] + l16) * KW;  // loop-invariant: L0/L2 hot
  }

  float c_st[8], h_st[8];
#pragma unroll
  for (int r = 0; r < 8; ++r) { c_st[r] = 0.0f; h_st[r] = 0.0f; }

  __syncthreads();

  for (int step = 0; step < Tsz; ++step) {
    const int t  = dir ? (Tsz - 1 - step) : step;
    const int rb = step & 1, wb = rb ^ 1;

    // stream this step's precomputed x-preactivations (C-fragment layout)
    const float* zt = Zd + (size_t)t * (Bsz * G4H);
    float zg[4][8];
#pragma unroll
    for (int g = 0; g < 4; ++g)
#pragma unroll
      for (int r = 0; r < 8; ++r)
        zg[g][r] = zt[(mt * 16 + hi * 8 + r) * G4H + ncol[g] + l16];

    // prefetch next step's Z segments (global_prefetch): lane j covers row
    // mt*16 + (j&15) of gate p -> 16 rows x 4 gates one step ahead of use.
    if (step + 1 < Tsz) {
      const float* ztn = Zd + (size_t)(dir ? t - 1 : t + 1) * (Bsz * G4H);
#pragma unroll
      for (int p = 0; p < 4; ++p)
        __builtin_prefetch(ztn + (mt * 16 + (lane & 15)) * G4H + ncol[p], 0, 0);
    }

    // recurrent GEMM: four 16x16 gate tiles, K = 256 in 8 WMMA chunks
    v8f acc0 = {}, acc1 = {}, acc2 = {}, acc3 = {};
    const __bf16* arow = &hbuf[rb][mt * 16 + l16][0];
#pragma unroll
    for (int kc = 0; kc < 8; ++kc) {
      const int koA = kc * 32 + hi * 16;
      const int koB = kc * 32 + hi * 16;
      v16bf a = ldA16(arow + koA);
      acc0 = wmma_bf16(a, ldB16(bptr[0] + koB), acc0);
      acc1 = wmma_bf16(a, ldB16(bptr[1] + koB), acc1);
      acc2 = wmma_bf16(a, ldB16(bptr[2] + koB), acc2);
      acc3 = wmma_bf16(a, ldB16(bptr[3] + koB), acc3);
    }

    // gate math (f32), length masking, register state update, output store
#pragma unroll
    for (int r = 0; r < 8; ++r) {
      const int  b   = mt * 16 + hi * 8 + r;
      const bool upd = t < lenr[r];
      const float zi = acc0[r] + zg[0][r];
      const float zj = acc1[r] + zg[1][r];
      const float zf = acc2[r] + zg[2][r];
      const float zo = acc3[r] + zg[3][r];
      const float cn = fsigmoid(zf + 1.0f) * c_st[r] + fsigmoid(zi) * ftanh_(zj);
      const float hn = fsigmoid(zo) * ftanh_(cn);
      out[((size_t)b * Tsz + t) * (2 * Hsz) + dir * Hsz + hcol] = upd ? hn : 0.0f;
      if (upd) { c_st[r] = cn; h_st[r] = hn; }
      hbuf[wb][b][hcol] = (__bf16)h_st[r];       // bf16 h for next step's A
    }
    __syncthreads();  // one barrier per timestep (double-buffered h)
  }
}

// ---------------------------------------------------------------------------
extern "C" void kernel_launch(void* const* d_in, const int* in_sizes, int n_in,
                              void* d_out, int out_size, void* d_ws, size_t ws_size,
                              hipStream_t stream) {
  (void)in_sizes; (void)n_in; (void)out_size; (void)ws_size;
  const float* seqs    = (const float*)d_in[0];
  const int*   lengths = (const int*)  d_in[1];
  const float* W_fwd   = (const float*)d_in[2];
  const float* b_fwd   = (const float*)d_in[3];
  const float* W_bwd   = (const float*)d_in[4];
  const float* b_bwd   = (const float*)d_in[5];
  float* out = (float*)d_out;

  char*   ws  = (char*)d_ws;
  __bf16* wbf = (__bf16*)(ws + WBF_OFF);
  __bf16* xbf = (__bf16*)(ws + XBF_OFF);
  float*  Z   = (float*)(ws + Z_OFF);

  // Phase 0: bf16 conversions
  lstm_prep_kernel<<<4096, 256, 0, stream>>>(seqs, W_fwd, W_bwd, xbf, wbf);
  // Phase 1: Z = X @ W_x^T + b for both directions (262144 WMMA tile jobs)
  lstm_zgemm_kernel<<<32768, 256, 0, stream>>>(xbf, wbf, b_fwd, b_bwd, Z);
  // Phase 2: sequential recurrence, one WGP-resident workgroup per direction
  lstm_seq_kernel<<<2, 1024, 0, stream>>>(wbf, Z, lengths, out);
}
